// ASDALayer_56057913147948
// MI455X (gfx1250) — compile-verified, matrix operand order checked by reading, not run on hardware
//
#include <hip/hip_runtime.h>
#include <hip/hip_bf16.h>
#include <hip/hip_fp16.h>

typedef __attribute__((ext_vector_type(16))) _Float16 v16h;
typedef __attribute__((ext_vector_type(8)))  float    v8f;

#define TAU_INV 10.0f

__device__ __forceinline__ void atomAddF(float* p, float v) {
  // lowers to global_atomic_add_f32 (relaxed, device scope)
  __hip_atomic_fetch_add(p, v, __ATOMIC_RELAXED, __HIP_MEMORY_SCOPE_AGENT);
}

// float atomic max via int-max / uint-min trick; dest must be initialized to
// bit pattern 0xFF800000 (-inf).
__device__ __forceinline__ void atomMaxF(float* p, float v) {
  if (v >= 0.0f) atomicMax((int*)p, __float_as_int(v));
  else           atomicMin((unsigned int*)p, __float_as_uint(v));
}

// |a| via an opaque v_and_b32: keeps InstCombine/SLP from re-forming the
// llvm.fabs pattern that crashes this LLVM build's gfx1250 isel.
__device__ __forceinline__ float fabs_opaque(float a) {
  float r;
  asm("v_and_b32 %0, 0x7fffffff, %1" : "=v"(r) : "v"(a));
  return r;
}

// ---------------------------------------------------------------- fill
__global__ void k_fill(unsigned int* __restrict__ p, unsigned int pat, int n) {
  int i = blockIdx.x * blockDim.x + threadIdx.x;
  if (i < n) p[i] = pat;
}

// ---------------------------------------------------------------- hsd stats
__global__ void k_stats(const float* __restrict__ hsd, float* __restrict__ stats, int n) {
  __shared__ float s0[256], s1[256];
  float a = 0.f, b = 0.f;
  for (int i = blockIdx.x * blockDim.x + threadIdx.x; i < n; i += gridDim.x * blockDim.x) {
    float h = hsd[i];
    a += h;
    b += h * h;
  }
  s0[threadIdx.x] = a; s1[threadIdx.x] = b;
  __syncthreads();
  for (int s = 128; s > 0; s >>= 1) {
    if (threadIdx.x < (unsigned)s) {
      s0[threadIdx.x] += s0[threadIdx.x + s];
      s1[threadIdx.x] += s1[threadIdx.x + s];
    }
    __syncthreads();
  }
  if (threadIdx.x == 0) { atomAddF(&stats[0], s0[0]); atomAddF(&stats[1], s1[0]); }
}

// ------------------------------------------------- per-node: norm + gate MLP
__global__ void k_node(const float* __restrict__ hsd, const float* __restrict__ stats,
                       const float* __restrict__ Wn1, const float* __restrict__ bn1,
                       const float* __restrict__ Wn2, const float* __restrict__ bn2,
                       float* __restrict__ hnorm, float* __restrict__ abar, int n) {
  int i = blockIdx.x * blockDim.x + threadIdx.x;
  if (i >= n) return;
  float mean = stats[0] / (float)n;
  float var  = (stats[1] - stats[0] * mean) / (float)(n - 1);  // ddof=1
  float sd   = sqrtf(fmaxf(var, 0.f));
  float hn   = (hsd[i] - mean) / (sd + 1e-8f);
  hnorm[i] = hn;
  float s = bn2[0];
#pragma unroll
  for (int j = 0; j < 16; ++j)
    s += fmaxf(fmaf(hn, Wn1[j], bn1[j]), 0.f) * Wn2[j];
  abar[i] = 1.f / (1.f + __expf(-s));
}

// -------------------------------- edge MLP via WMMA: 16 edges per wave/tile
// Full 34-dim input handled as two chained WMMAs per output half:
//   C  = be1                                  (bias)
//   C += [hu,hv,0..] (16x32) @ We1[0:2,:]     (rank-2 hsd term, K=0,1 only)
//   C += |x_u-x_v|   (16x32) @ We1[2:34,:]    (feature term)
__global__ void __launch_bounds__(256)
k_edge_mlp(const float* __restrict__ x,
           const int* __restrict__ erow, const int* __restrict__ ecol,
           const float* __restrict__ hnorm,
           const float* __restrict__ We1, const float* __restrict__ be1,
           const float* __restrict__ We2, const float* __restrict__ be2,
           float* __restrict__ logits, float* __restrict__ segmax,
           float* __restrict__ degs, float* __restrict__ degd,
           int ne, int ntiles) {
  const int lane = threadIdx.x & 31;
  const int cn   = lane & 15;   // A-row / B,C,D column index for this lane
  const int kh   = lane >> 4;   // K-half select (ISA 16-bit A/B layout)
  const int wave = blockIdx.x * (blockDim.x >> 5) + (threadIdx.x >> 5);
  const int nwav = gridDim.x * (blockDim.x >> 5);

  // Feature B tiles: We1 rows 2..33 (32x32), split into output cols [0,16)/[16,32).
  v16h b0, b1;
#pragma unroll
  for (int j = 0; j < 8; ++j) {
    int k0 = 8 * kh + j;
    int k1 = 16 + 8 * kh + j;
    b0[j]     = (_Float16)We1[(2 + k0) * 32 + cn];
    b0[j + 8] = (_Float16)We1[(2 + k1) * 32 + cn];
    b1[j]     = (_Float16)We1[(2 + k0) * 32 + 16 + cn];
    b1[j + 8] = (_Float16)We1[(2 + k1) * 32 + 16 + cn];
  }
  // hsd B tiles: We1 rows 0..1 in K=0,1 slots (held by kh==0 lanes), rest zero.
  v16h bh0 = {}, bh1 = {};
  bh0[0] = (_Float16)(kh == 0 ? We1[cn] : 0.f);          // K=0 row, col cn
  bh0[1] = (_Float16)(kh == 0 ? We1[32 + cn] : 0.f);     // K=1 row, col cn
  bh1[0] = (_Float16)(kh == 0 ? We1[16 + cn] : 0.f);     // K=0 row, col 16+cn
  bh1[1] = (_Float16)(kh == 0 ? We1[48 + cn] : 0.f);     // K=1 row, col 16+cn

  const float bea = be1[cn],  beb = be1[16 + cn];
  const float w2a = We2[cn],  w2b = We2[16 + cn];
  const float be2s = be2[0];

  for (int t = wave; t < ntiles; t += nwav) {
    const int base = t * 16;
    const int e = min(base + cn, ne - 1);   // this lane's edge (dup across halves)
    const int u = erow[e], v = ecol[e];
    const float hu = hnorm[u], hv = hnorm[v];

    // hsd A tile: row m=cn has [hu,hv] at K=0,1 (owned by kh==0 lanes).
    v16h A2 = {};
    A2[0] = (_Float16)(kh == 0 ? hu : 0.f);
    A2[1] = (_Float16)(kh == 0 ? hv : 0.f);

    // Feature A tile: |x_u - x_v|, two contiguous 8-float chunks per lane
    // chunk0: feats [8*kh, 8*kh+8), chunk1: feats [16+8*kh, 16+8*kh+8)
    const float4* xu4 = (const float4*)(x + (size_t)u * 32);
    const float4* xv4 = (const float4*)(x + (size_t)v * 32);
    float4 a0 = xu4[2 * kh + 0], a1 = xu4[2 * kh + 1];
    float4 a2 = xu4[2 * kh + 4], a3 = xu4[2 * kh + 5];
    float4 p0 = xv4[2 * kh + 0], p1 = xv4[2 * kh + 1];
    float4 p2 = xv4[2 * kh + 4], p3 = xv4[2 * kh + 5];

    v16h A;
    A[0]  = (_Float16)fabs_opaque(a0.x - p0.x); A[1]  = (_Float16)fabs_opaque(a0.y - p0.y);
    A[2]  = (_Float16)fabs_opaque(a0.z - p0.z); A[3]  = (_Float16)fabs_opaque(a0.w - p0.w);
    A[4]  = (_Float16)fabs_opaque(a1.x - p1.x); A[5]  = (_Float16)fabs_opaque(a1.y - p1.y);
    A[6]  = (_Float16)fabs_opaque(a1.z - p1.z); A[7]  = (_Float16)fabs_opaque(a1.w - p1.w);
    A[8]  = (_Float16)fabs_opaque(a2.x - p2.x); A[9]  = (_Float16)fabs_opaque(a2.y - p2.y);
    A[10] = (_Float16)fabs_opaque(a2.z - p2.z); A[11] = (_Float16)fabs_opaque(a2.w - p2.w);
    A[12] = (_Float16)fabs_opaque(a3.x - p3.x); A[13] = (_Float16)fabs_opaque(a3.y - p3.y);
    A[14] = (_Float16)fabs_opaque(a3.z - p3.z); A[15] = (_Float16)fabs_opaque(a3.w - p3.w);

    // C = bias; += hsd rank-2 term; += feature term (all on the matrix unit)
    v8f c0, c1;
#pragma unroll
    for (int r = 0; r < 8; ++r) { c0[r] = bea; c1[r] = beb; }

    c0 = __builtin_amdgcn_wmma_f32_16x16x32_f16(false, A2, false, bh0, (short)0, c0, false, false);
    c1 = __builtin_amdgcn_wmma_f32_16x16x32_f16(false, A2, false, bh1, (short)0, c1, false, false);
    c0 = __builtin_amdgcn_wmma_f32_16x16x32_f16(false, A,  false, b0,  (short)0, c0, false, false);
    c1 = __builtin_amdgcn_wmma_f32_16x16x32_f16(false, A,  false, b1,  (short)0, c1, false, false);

    // relu + dot with We2: butterfly-reduce each C row across its 16 lanes
    float pr[8];
#pragma unroll
    for (int r = 0; r < 8; ++r)
      pr[r] = fmaxf(c0[r], 0.f) * w2a + fmaxf(c1[r], 0.f) * w2b;
#pragma unroll
    for (int m = 1; m < 16; m <<= 1) {
#pragma unroll
      for (int r = 0; r < 8; ++r) pr[r] += __shfl_xor(pr[r], m);
    }

    // writers: lane cn==r in each half writes row 8*kh+r
#pragma unroll
    for (int r = 0; r < 8; ++r) {
      if (cn == r) {
        int rowi = base + 8 * kh + r;
        if (rowi < ne) {
          float lg = (pr[r] + be2s) * TAU_INV;
          logits[rowi] = lg;
          atomMaxF(&segmax[ecol[rowi]], lg);
        }
      }
    }
    // degree counts (one half only, one lane per edge)
    if (kh == 0 && base + cn < ne) {
      atomAddF(&degs[u], 1.f);
      atomAddF(&degd[v], 1.f);
    }
  }
}

// ------------------------------------------ exp(logit - segmax) + denom sums
__global__ void k_edge_exp(const int* __restrict__ ecol, const float* __restrict__ segmax,
                           float* __restrict__ exbuf, float* __restrict__ denom, int ne) {
  int e = blockIdx.x * blockDim.x + threadIdx.x;
  if (e >= ne) return;
  int v = ecol[e];
  float m = segmax[v];
  if (!(m > -1e30f && m < 1e30f)) m = 0.f;   // mimic isfinite guard
  float ex = __expf(exbuf[e] - m);
  exbuf[e] = ex;
  atomAddF(&denom[v], ex);
}

// --------------------------- per-edge scatter: 8 threads x float4 per edge
__global__ void k_edge_scatter(const float* __restrict__ x,
                               const int* __restrict__ erow, const int* __restrict__ ecol,
                               const float* __restrict__ exbuf, const float* __restrict__ denom,
                               const float* __restrict__ degs, const float* __restrict__ degd,
                               float* __restrict__ lowp, float* __restrict__ highp, int ne) {
  long long gid = (long long)blockIdx.x * blockDim.x + threadIdx.x;
  int e  = (int)(gid >> 3);
  int ch = (int)(gid & 7);
  if (e >= ne) return;
  int u = erow[e], v = ecol[e];
  float alpha = exbuf[e] / (denom[v] + 1e-16f);
  float w = rsqrtf(fmaxf(degs[u], 1.f)) * rsqrtf(fmaxf(degd[v], 1.f));
  const float4 xu = ((const float4*)x)[(size_t)u * 8 + ch];
  const float4 xv = ((const float4*)x)[(size_t)v * 8 + ch];
  float* lp = lowp  + (size_t)v * 32 + ch * 4;
  float* hp = highp + (size_t)v * 32 + ch * 4;
  atomAddF(lp + 0, w * xu.x); atomAddF(lp + 1, w * xu.y);
  atomAddF(lp + 2, w * xu.z); atomAddF(lp + 3, w * xu.w);
  atomAddF(hp + 0, alpha * (xv.x - xu.x)); atomAddF(hp + 1, alpha * (xv.y - xu.y));
  atomAddF(hp + 2, alpha * (xv.z - xu.z)); atomAddF(hp + 3, alpha * (xv.w - xu.w));
}

// ---------------------------------------------------------------- blend out
__global__ void k_out(const float* __restrict__ lowp, const float* __restrict__ highp,
                      const float* __restrict__ abar, float* __restrict__ out, int total) {
  int i = blockIdx.x * blockDim.x + threadIdx.x;
  if (i >= total) return;
  float ab = abar[i >> 5];
  out[i] = (1.f - ab) * lowp[i] + ab * highp[i];
}

extern "C" void kernel_launch(void* const* d_in, const int* in_sizes, int n_in,
                              void* d_out, int out_size, void* d_ws, size_t ws_size,
                              hipStream_t stream) {
  const float* x   = (const float*)d_in[0];
  const float* hsd = (const float*)d_in[1];
  const float* We1 = (const float*)d_in[2];
  const float* be1 = (const float*)d_in[3];
  const float* We2 = (const float*)d_in[4];
  const float* be2 = (const float*)d_in[5];
  const float* Wn1 = (const float*)d_in[6];
  const float* bn1 = (const float*)d_in[7];
  const float* Wn2 = (const float*)d_in[8];
  const float* bn2 = (const float*)d_in[9];
  const int*   ei  = (const int*)d_in[10];

  const int N = in_sizes[1];
  const int E = in_sizes[10] / 2;
  const int* erow = ei;
  const int* ecol = ei + E;

  // workspace layout (floats):
  // [stats 8][segmax N][denom N][degs N][degd N][low 32N][high 32N][hnorm N][abar N][ex E]
  float* ws     = (float*)d_ws;
  float* stats  = ws;
  float* segmax = stats + 8;
  float* denom  = segmax + N;
  float* degs   = denom + N;
  float* degd   = degs + N;
  float* lowp   = degd + N;
  float* highp  = lowp + (size_t)32 * N;
  float* hnorm  = highp + (size_t)32 * N;
  float* abar   = hnorm + N;
  float* exbuf  = abar + N;
  float* outp   = (float*)d_out;

  // init: stats=0, segmax=-inf, [denom..high]=0 (contiguous 67N floats)
  k_fill<<<1, 256, 0, stream>>>((unsigned int*)stats, 0u, 8);
  k_fill<<<(N + 255) / 256, 256, 0, stream>>>((unsigned int*)segmax, 0xFF800000u, N);
  int zcnt = 67 * N;
  k_fill<<<(zcnt + 255) / 256, 256, 0, stream>>>((unsigned int*)denom, 0u, zcnt);

  k_stats<<<512, 256, 0, stream>>>(hsd, stats, N);
  k_node<<<(N + 255) / 256, 256, 0, stream>>>(hsd, stats, Wn1, bn1, Wn2, bn2, hnorm, abar, N);

  int ntiles = (E + 15) / 16;
  k_edge_mlp<<<1600, 256, 0, stream>>>(x, erow, ecol, hnorm, We1, be1, We2, be2,
                                       exbuf, segmax, degs, degd, E, ntiles);

  k_edge_exp<<<(E + 255) / 256, 256, 0, stream>>>(ecol, segmax, exbuf, denom, E);

  long long sg = (long long)E * 8;
  k_edge_scatter<<<(int)((sg + 255) / 256), 256, 0, stream>>>(x, erow, ecol, exbuf, denom,
                                                              degs, degd, lowp, highp, E);

  int total = N * 32;
  k_out<<<(total + 255) / 256, 256, 0, stream>>>(lowp, highp, abar, outp, total);
}
